// QuantLinear_74148315398648
// MI455X (gfx1250) — matrix operand order repaired and checked by reading, not compiled, not run on hardware
//
#include <hip/hip_runtime.h>

// ---------------------------------------------------------------------------
// GPTQ 4-bit quantized linear for MI455X (gfx1250, wave32, WMMA).
// out[256,11008] = x[256,4096] @ dequant(W) + bias, groupsize 128.
// Compute-bound (577 FLOP/byte) -> f16 WMMA with fp32 accumulate.
// A tile staged with GLOBAL_LOAD_ASYNC_TO_LDS_B128 (ASYNCcnt),
// qweight loads software-pipelined under the WMMA phase.
// ---------------------------------------------------------------------------

typedef __attribute__((ext_vector_type(16))) _Float16 v16h;
typedef __attribute__((ext_vector_type(8)))  _Float16 f16x8;
typedef __attribute__((ext_vector_type(4)))  _Float16 f16x4;
typedef __attribute__((ext_vector_type(2)))  _Float16 f16x2;
typedef __attribute__((ext_vector_type(8)))  float    v8f;

#define NTOK 256
#define INF  4096
#define OUTF 11008
#define BM   128   // M tile per block
#define BN   128   // N tile per block
#define KB   64    // K staged per LDS iteration (2 WMMA k-steps)
#define LDA  72    // As row stride in halves (KB + 8 pad, keeps 16B alignment)
#define LDB  136   // Bs row stride in halves (BN + 8 pad, keeps 16B alignment)

// ---- one-shot fp32 -> fp16 conversion of x into workspace (x is L2-resident,
//      halving its element size halves per-block re-read traffic) -------------
__global__ __launch_bounds__(256) void convert_x_kernel(const float* __restrict__ x,
                                                        _Float16* __restrict__ xh) {
  const int i = blockIdx.x * blockDim.x + threadIdx.x;   // 0 .. NTOK*INF/4-1
  const float4 v = ((const float4*)x)[i];
  f16x4 h;
  h[0] = (_Float16)v.x; h[1] = (_Float16)v.y;
  h[2] = (_Float16)v.z; h[3] = (_Float16)v.w;
  *(f16x4*)(xh + (size_t)i * 4) = h;
}

// ---- main WMMA GEMM with fused 4-bit dequant --------------------------------
__global__ __launch_bounds__(256) void gptq_wmma_kernel(
    const _Float16* __restrict__ xh,      // [NTOK][INF] f16
    const float*    __restrict__ scales,  // [32][OUTF]
    const float*    __restrict__ bias,    // [OUTF]
    const int*      __restrict__ qweight, // [INF/8][OUTF], 8 K-vals per i32
    const int*      __restrict__ qzeros,  // [32][OUTF/8],  8 N-vals per i32
    float*          __restrict__ out) {   // [NTOK][OUTF]
  __shared__ alignas(16) _Float16 As[BM][LDA];
  __shared__ alignas(16) _Float16 Bs[KB][LDB];

  const int n_tile = blockIdx.x * BN;
  const int m_tile = blockIdx.y * BM;
  const int t    = threadIdx.x;
  const int lane = t & 31;
  const int wave = t >> 5;
  const int wm   = (wave & 1) * 64;   // wave's M offset inside tile
  const int wn   = (wave >> 1) * 32;  // wave's N offset inside tile

  // --- B-producer mapping: each thread dequants 2 adjacent columns x 2 packed
  //     rows (= 2 cols x 16 K values) per K-iteration ---
  const int nloc  = (t & 63) * 2;       // local column pair: 0,2,...,126
  const int rb    = (t >> 6) * 2;       // packed-row base: 0,2,4,6
  const int nglob = n_tile + nloc;
  const int zsh   = 4 * (nglob & 7);    // nglob even -> pair shares one i32
  const int*   qz_col = qzeros + (nglob >> 3);
  const float* sc_col = scales + nglob;
  const int*   qw_col = qweight + nglob;

  // --- A-producer mapping: 4 async 16B chunks per thread per iteration ---
  int a_row[4], a_ch[4];
  unsigned a_lds[4];
#pragma unroll
  for (int p = 0; p < 4; ++p) {
    const int idx = p * 256 + t;         // 1024 16-byte chunks per tile
    a_row[p] = idx >> 3;                 // 0..127
    a_ch[p]  = (idx & 7) * 8;            // half offset 0,8,...,56
    // Generic LDS address: addr[31:0] IS the LDS byte offset on gfx1250.
    a_lds[p] = (unsigned)(size_t)&As[a_row[p]][a_ch[p]];
  }

  v8f acc[4][2];
  {
    const v8f z = {};
#pragma unroll
    for (int i = 0; i < 4; ++i)
#pragma unroll
      for (int j = 0; j < 2; ++j) acc[i][j] = z;
  }

  // Pipelined packed-weight registers for the first K tile.
  int2 wq[2];
#pragma unroll
  for (int r = 0; r < 2; ++r)
    wq[r] = *(const int2*)(qw_col + (size_t)(rb + r) * OUTF);

  for (int k0 = 0; k0 < INF; k0 += KB) {
    // ------- A tile: async DMA global f16 -> LDS (no VGPR data movement) ----
#pragma unroll
    for (int p = 0; p < 4; ++p) {
      const unsigned goff =
          (unsigned)(((m_tile + a_row[p]) * INF + k0 + a_ch[p]) * 2);
      asm volatile("global_load_async_to_lds_b128 %0, %1, %2"
                   :: "v"(a_lds[p]), "v"(goff), "s"(xh)
                   : "memory");
    }

    // ------- B tile: dequant pipelined registers -> LDS ---------------------
    {
      const int   g  = k0 >> 7;          // groupsize 128, KB=64 -> one group/tile
      const int   zw = qz_col[g * (OUTF / 8)];
      const float z0 = (float)(((zw >> zsh) & 15) + 1);
      const float z1 = (float)(((zw >> (zsh + 4)) & 15) + 1);
      const float s0 = sc_col[(size_t)g * OUTF];
      const float s1 = sc_col[(size_t)g * OUTF + 1];
      const float c0 = -s0 * z0;         // fold into single FMA per value
      const float c1 = -s1 * z1;
#pragma unroll
      for (int r = 0; r < 2; ++r) {
        const int  prow = rb + r;        // packed row 0..7  (K = k0 + prow*8 + j)
        const int2 w = wq[r];
#pragma unroll
        for (int j = 0; j < 8; ++j) {
          const float w0 = (float)((w.x >> (4 * j)) & 15) * s0 + c0;
          const float w1 = (float)((w.y >> (4 * j)) & 15) * s1 + c1;
          f16x2 hv; hv[0] = (_Float16)w0; hv[1] = (_Float16)w1;
          *(f16x2*)&Bs[prow * 8 + j][nloc] = hv;
        }
      }
    }

    asm volatile("s_wait_asynccnt 0x0" ::: "memory");  // A tile landed in LDS
    __syncthreads();

    // ------- issue next iteration's qweight loads (hide under WMMAs) --------
    if (k0 + KB < INF) {
#pragma unroll
      for (int r = 0; r < 2; ++r)
        wq[r] = *(const int2*)(qw_col + (size_t)((k0 + KB) / 8 + rb + r) * OUTF);
    }

    // ------- consume: 2 k-steps of v_wmma_f32_16x16x32_f16 ------------------
    {
      const int lm = lane & 15;
      const int lk = (lane >> 4) * 8;    // half-wave K offset (in halves)
#pragma unroll
      for (int ks = 0; ks < KB; ks += 32) {
        v16h afrag[4];
#pragma unroll
        for (int i = 0; i < 4; ++i) {
          // A layout: lanes 0-15 -> K {0..7, 16..23}; lanes 16-31 -> K {8..15, 24..31}
          const _Float16* ap = &As[wm + i * 16 + lm][ks + lk];
          const f16x8 lo = *(const f16x8*)ap;
          const f16x8 hi = *(const f16x8*)(ap + 16);
          afrag[i] = __builtin_shufflevector(lo, hi, 0, 1, 2, 3, 4, 5, 6, 7,
                                             8, 9, 10, 11, 12, 13, 14, 15);
        }
        v16h bfrag[2];
#pragma unroll
        for (int j = 0; j < 2; ++j) {
          // B layout: lane L holds column N=L%16; lanes 0-15 -> K {0..7, 16..23}; lanes 16-31 -> K {8..15, 24..31}
          const _Float16* bp = &Bs[ks + lk][wn + j * 16 + lm];
#pragma unroll
          for (int s = 0; s < 8; ++s) {
            bfrag[j][s]     = bp[s * LDB];
            bfrag[j][8 + s] = bp[(16 + s) * LDB];
          }
        }
#pragma unroll
        for (int i = 0; i < 4; ++i)
#pragma unroll
          for (int j = 0; j < 2; ++j)
            acc[i][j] = __builtin_amdgcn_wmma_f32_16x16x32_f16(
                false, afrag[i], false, bfrag[j], (short)0, acc[i][j],
                false, false);
      }
    }
    __syncthreads();   // protect LDS before next iteration's producers
  }

  // ---------------- epilogue: add bias, store per C/D VGPR layout ----------
  {
    const int lm = lane & 15;     // N within 16-subtile
    const int lh = lane >> 4;     // 0 -> M rows v, 1 -> M rows v+8
#pragma unroll
    for (int j = 0; j < 2; ++j) {
      const int ng = n_tile + wn + j * 16 + lm;
      const float bj = bias[ng];
#pragma unroll
      for (int i = 0; i < 4; ++i) {
#pragma unroll
        for (int v = 0; v < 8; ++v) {
          const int mg = m_tile + wm + i * 16 + lh * 8 + v;
          out[(size_t)mg * OUTF + ng] = acc[i][j][v] + bj;
        }
      }
    }
  }
}

extern "C" void kernel_launch(void* const* d_in, const int* in_sizes, int n_in,
                              void* d_out, int out_size, void* d_ws, size_t ws_size,
                              hipStream_t stream) {
  const float* x       = (const float*)d_in[0];
  const float* scales  = (const float*)d_in[1];
  const float* bias    = (const float*)d_in[2];
  const int*   qweight = (const int*)d_in[3];
  const int*   qzeros  = (const int*)d_in[4];
  // d_in[5] (g_idx) is the trivial k/128 grouping; handled analytically.
  float* out = (float*)d_out;
  _Float16* xh = (_Float16*)d_ws;   // needs NTOK*INF*2 = 2 MiB scratch

  convert_x_kernel<<<(NTOK * INF / 4) / 256, 256, 0, stream>>>(x, xh);

  dim3 grid(OUTF / BN, NTOK / BM);  // 86 x 2 blocks, 8 wave32 each
  gptq_wmma_kernel<<<grid, 256, 0, stream>>>(xh, scales, bias, qweight, qzeros, out);
}
